// GraphGenerator_63952063038068
// MI455X (gfx1250) — matrix operand order repaired and checked by reading, not compiled, not run on hardware
//
#include <hip/hip_runtime.h>
#include <hip/hip_bf16.h>

typedef __attribute__((ext_vector_type(16))) _Float16 v16h;
typedef __attribute__((ext_vector_type(8)))  float    v8f;

// ---------------------------------------------------------------------------
// WMMA helpers (CDNA5: D = A(16x32 f16) * B(32x16 f16) + C(16x16 f32))
// ---------------------------------------------------------------------------
__device__ __forceinline__ v8f wmma_f16(v16h a, v16h b, v8f c) {
  return __builtin_amdgcn_wmma_f32_16x16x32_f16(
      /*neg_a=*/false, a, /*neg_b=*/false, b,
      /*c_mod=*/(short)0, c, /*reuse_a=*/false, /*reuse_b=*/false);
}

// B tile (32x16, K x N) from row-major weight W[Kdim x Ndim], columns n0..n0+15.
// Layout: lane%16 selects N; element j -> K = j + 8*(lane>=16) + 8*(j>=8)
// (matches 16-bit A striping).
__device__ __forceinline__ v16h load_B_tile(const float* __restrict__ W,
                                            int Kdim, int Ndim, int n0, int lane) {
  v16h b;
  int n  = n0 + (lane & 15);
  int kb = (lane & 16) ? 8 : 0;
#pragma unroll
  for (int j = 0; j < 16; ++j) {
    int k = kb + j + (j >= 8 ? 8 : 0);
    float v = (k < Kdim && n < Ndim) ? W[k * Ndim + n] : 0.0f;
    b[j] = (_Float16)v;
  }
  return b;
}

// Store relu(C + bias) into per-wave LDS staging (16 edges x 32 channels f32).
__device__ __forceinline__ void stage_relu(float* __restrict__ hsw,
                                           const v8f& c0, const v8f& c1,
                                           float biasLo, float biasHi, int lane) {
  int n  = lane & 15;
  int mb = (lane & 16) ? 8 : 0;
#pragma unroll
  for (int r = 0; r < 8; ++r) {
    float vlo = c0[r] + biasLo; vlo = vlo > 0.f ? vlo : 0.f;
    float vhi = c1[r] + biasHi; vhi = vhi > 0.f ? vhi : 0.f;
    hsw[(mb + r) * 32 + n]      = vlo;
    hsw[(mb + r) * 32 + 16 + n] = vhi;
  }
}

// Rebuild A tile (16x32 f16) from staged LDS. A layout: lane%16 = M (edge),
// element j -> K = j + 8*(lane>=16) + 8*(j>=8)  (per ISA 16-bit A striping).
__device__ __forceinline__ v16h load_A_lds(const float* __restrict__ hsw, int lane) {
  int m  = lane & 15;
  int kb = (lane & 16) ? 8 : 0;
  const float4* row = reinterpret_cast<const float4*>(hsw + m * 32);
  float4 p0 = row[kb / 4];            // K = kb .. kb+3
  float4 p1 = row[kb / 4 + 1];        // K = kb+4 .. kb+7
  float4 p2 = row[(16 + kb) / 4];     // K = 16+kb ..
  float4 p3 = row[(16 + kb) / 4 + 1];
  v16h a;
  a[0]  = (_Float16)p0.x; a[1]  = (_Float16)p0.y; a[2]  = (_Float16)p0.z; a[3]  = (_Float16)p0.w;
  a[4]  = (_Float16)p1.x; a[5]  = (_Float16)p1.y; a[6]  = (_Float16)p1.z; a[7]  = (_Float16)p1.w;
  a[8]  = (_Float16)p2.x; a[9]  = (_Float16)p2.y; a[10] = (_Float16)p2.z; a[11] = (_Float16)p2.w;
  a[12] = (_Float16)p3.x; a[13] = (_Float16)p3.y; a[14] = (_Float16)p3.z; a[15] = (_Float16)p3.w;
  return a;
}

#define WAIT_DS() asm volatile("s_wait_dscnt 0" ::: "memory")

// ---------------------------------------------------------------------------
// Kernel A: xw = [x,token] @ Wc  and  deg init (self-loop weight = 1)
// ---------------------------------------------------------------------------
__global__ void __launch_bounds__(256) node_init_kernel(
    const float* __restrict__ x, const float* __restrict__ token,
    const float* __restrict__ Wc, float* __restrict__ xw,
    float* __restrict__ deg, int N) {
  int i = blockIdx.x * blockDim.x + threadIdx.x;
  if (i >= N) return;
  float xa[6];
  xa[0] = x[i];
#pragma unroll
  for (int f = 1; f < 6; ++f) xa[f] = token[i * 5 + f - 1];
#pragma unroll
  for (int c = 0; c < 10; ++c) {
    float acc = 0.f;
#pragma unroll
    for (int f = 0; f < 6; ++f) acc += xa[f] * Wc[f * 10 + c];
    xw[i * 10 + c] = acc;
  }
  deg[i] = 1.0f;
}

// ---------------------------------------------------------------------------
// Kernel B: per-edge MLP via WMMA, gumbel argmax, edge weight, deg scatter.
// Each wave handles 16 edges per iteration (one A tile).
// ---------------------------------------------------------------------------
__global__ void __launch_bounds__(256) edge_mlp_kernel(
    const float* __restrict__ x, const float* __restrict__ token,
    const float* __restrict__ eag, const float* __restrict__ eav,
    const float* __restrict__ gum, const int* __restrict__ eidx,
    const float* __restrict__ W0, const float* __restrict__ b0,
    const float* __restrict__ W1, const float* __restrict__ b1,
    const float* __restrict__ W2, const float* __restrict__ b2,
    const float* __restrict__ W3, const float* __restrict__ b3,
    const float* __restrict__ Wl, const float* __restrict__ bl,
    float* __restrict__ ew_out, float* __restrict__ deg, int E) {
  __shared__ __align__(16) float hs[8 * 16 * 32];   // 2KB staging per wave
  __shared__ float sW3[60];
  __shared__ float sb3[2];
  __shared__ float sWlbl[2];

  int tid = threadIdx.x;
  if (tid < 60) sW3[tid] = W3[tid];
  if (tid == 64) { sb3[0] = b3[0]; sb3[1] = b3[1]; sWlbl[0] = Wl[0]; sWlbl[1] = bl[0]; }
  __syncthreads();

  int lane = tid & 31;
  int wave = tid >> 5;
  float* hsw = hs + wave * (16 * 32);

  // Weight tiles: built once per wave, reused across all edge groups.
  v16h B0a = load_B_tile(W0, 13, 30, 0, lane);
  v16h B0b = load_B_tile(W0, 13, 30, 16, lane);
  v16h B1a = load_B_tile(W1, 30, 30, 0, lane);
  v16h B1b = load_B_tile(W1, 30, 30, 16, lane);
  v16h B2a = load_B_tile(W2, 30, 30, 0, lane);
  v16h B2b = load_B_tile(W2, 30, 30, 16, lane);

  int n = lane & 15;
  float bias0lo = b0[n], bias0hi = (n + 16 < 30) ? b0[n + 16] : 0.f;
  float bias1lo = b1[n], bias1hi = (n + 16 < 30) ? b1[n + 16] : 0.f;
  float bias2lo = b2[n], bias2hi = (n + 16 < 30) ? b2[n + 16] : 0.f;

  v8f zf = {};
  int ngroups = (E + 15) >> 4;
  int gwave   = blockIdx.x * 8 + wave;
  int gstride = gridDim.x * 8;

  for (int g = gwave; g < ngroups; g += gstride) {   // wave-uniform loop (EXEC full)
    int m  = lane & 15;
    int eo = g * 16 + m;
    int e  = eo < E ? eo : E - 1;                    // clamp, no EXEC change
    int s  = eidx[e];
    int t  = eidx[E + e];

    // A tile: 13 edge features, K-padded to 32 with zeros.
    v16h a = {};
    if (lane < 16) {                                 // K = 0..7
      a[0] = (_Float16)x[s];
      a[1] = (_Float16)token[s * 5 + 0];
      a[2] = (_Float16)token[s * 5 + 1];
      a[3] = (_Float16)token[s * 5 + 2];
      a[4] = (_Float16)token[s * 5 + 3];
      a[5] = (_Float16)token[s * 5 + 4];
      a[6] = (_Float16)x[t];
      a[7] = (_Float16)token[t * 5 + 0];
    } else {                                         // K = 8..15
      a[0] = (_Float16)token[t * 5 + 1];
      a[1] = (_Float16)token[t * 5 + 2];
      a[2] = (_Float16)token[t * 5 + 3];
      a[3] = (_Float16)token[t * 5 + 4];
      a[4] = (_Float16)eag[e];
    }

    v8f c0 = wmma_f16(a, B0a, zf);
    v8f c1 = wmma_f16(a, B0b, zf);
    stage_relu(hsw, c0, c1, bias0lo, bias0hi, lane);
    WAIT_DS();
    a  = load_A_lds(hsw, lane);
    c0 = wmma_f16(a, B1a, zf);
    c1 = wmma_f16(a, B1b, zf);
    stage_relu(hsw, c0, c1, bias1lo, bias1hi, lane);
    WAIT_DS();
    a  = load_A_lds(hsw, lane);
    c0 = wmma_f16(a, B2a, zf);
    c1 = wmma_f16(a, B2b, zf);
    stage_relu(hsw, c0, c1, bias2lo, bias2hi, lane);
    WAIT_DS();

    // Final 30->2 layer: VALU dot products from staged row (cheaper than WMMA).
    const float* hrow = hsw + m * 32;
    float l0 = sb3[0], l1 = sb3[1];
#pragma unroll
    for (int k = 0; k < 30; ++k) {
      float hv = hrow[k];
      l0 += hv * sW3[2 * k];
      l1 += hv * sW3[2 * k + 1];
    }
    // Straight-through gumbel: forward value == one_hot(argmax(logits+gumbel)).
    float z0 = l0 + gum[2 * e];
    float z1 = l1 + gum[2 * e + 1];
    float ev = (z1 > z0) ? 1.f : 0.f;                // edge_value = y_hard[:,1]
    float a0 = eav[2 * e];
    float mask = a0 * (1.f - ev);
    float wgt  = mask * (1.f / (1.f + __expf(-(a0 * sWlbl[0] + sWlbl[1]))));

    if (lane < 16 && eo < E) {                       // one lane per edge
      ew_out[eo] = wgt;
      if (wgt != 0.f) atomicAdd(&deg[t], wgt);
    }
  }
}

// ---------------------------------------------------------------------------
// Kernel C: dis = rsqrt(deg); self-loop contribution agg = xw / deg
// (deg array is overwritten with dis for kernel D)
// ---------------------------------------------------------------------------
__global__ void __launch_bounds__(256) node_norm_kernel(
    float* __restrict__ deg_dis, const float* __restrict__ xw,
    float* __restrict__ agg, int N) {
  int i = blockIdx.x * blockDim.x + threadIdx.x;
  if (i >= N) return;
  float d   = deg_dis[i];
  float dis = d > 0.f ? rsqrtf(d) : 0.f;
  float inv = dis * dis;                             // self-loop norm = dis^2
  deg_dis[i] = dis;
#pragma unroll
  for (int c = 0; c < 10; ++c) agg[i * 10 + c] = xw[i * 10 + c] * inv;
}

// ---------------------------------------------------------------------------
// Kernel D: agg[tgt] += dis[src]*ew*dis[tgt] * xw[src]   (10-ch scatter)
// ---------------------------------------------------------------------------
__global__ void __launch_bounds__(256) edge_agg_kernel(
    const int* __restrict__ eidx, const float* __restrict__ ew,
    const float* __restrict__ dis, const float* __restrict__ xw,
    float* __restrict__ agg, int E) {
  int e = blockIdx.x * blockDim.x + threadIdx.x;
  if (e >= E) return;
  float w = ew[e];
  if (w == 0.f) return;                              // argmax==1 edges are masked
  int s = eidx[e];
  int t = eidx[E + e];
  float nm = dis[s] * w * dis[t];
#pragma unroll
  for (int c = 0; c < 10; ++c) atomicAdd(&agg[t * 10 + c], nm * xw[s * 10 + c]);
}

// ---------------------------------------------------------------------------
// Kernel E: out = sigmoid((agg + bc) @ Wo + bo)
// ---------------------------------------------------------------------------
__global__ void __launch_bounds__(256) node_out_kernel(
    const float* __restrict__ agg, const float* __restrict__ bc,
    const float* __restrict__ Wo, const float* __restrict__ bo,
    float* __restrict__ out, int N) {
  int i = blockIdx.x * blockDim.x + threadIdx.x;
  if (i >= N) return;
  float acc = bo[0];
#pragma unroll
  for (int c = 0; c < 10; ++c) acc += (agg[i * 10 + c] + bc[c]) * Wo[c];
  out[i] = 1.f / (1.f + __expf(-acc));
}

// ---------------------------------------------------------------------------
extern "C" void kernel_launch(void* const* d_in, const int* in_sizes, int n_in,
                              void* d_out, int out_size, void* d_ws, size_t ws_size,
                              hipStream_t stream) {
  const float* x     = (const float*)d_in[0];
  const float* token = (const float*)d_in[1];
  const float* eag   = (const float*)d_in[2];
  const float* eav   = (const float*)d_in[3];
  const float* gum   = (const float*)d_in[4];
  const int*   eidx  = (const int*)d_in[5];
  const float* W0 = (const float*)d_in[6];  const float* b0 = (const float*)d_in[7];
  const float* W1 = (const float*)d_in[8];  const float* b1 = (const float*)d_in[9];
  const float* W2 = (const float*)d_in[10]; const float* b2 = (const float*)d_in[11];
  const float* W3 = (const float*)d_in[12]; const float* b3 = (const float*)d_in[13];
  const float* Wl = (const float*)d_in[14]; const float* bl = (const float*)d_in[15];
  const float* Wc = (const float*)d_in[16]; const float* bc = (const float*)d_in[17];
  const float* Wo = (const float*)d_in[18]; const float* bo = (const float*)d_in[19];

  int N = in_sizes[0];   // x has N elements
  int E = in_sizes[2];   // edge_attr_gen has E elements
  float* out = (float*)d_out;

  // Workspace layout (floats): ew[E] | deg/dis[N] | xw[10N] | agg[10N]
  float* ws  = (float*)d_ws;
  float* ew  = ws;
  float* deg = ws + (size_t)E;
  float* xw  = ws + (size_t)E + (size_t)N;
  float* agg = ws + (size_t)E + (size_t)11 * N;

  int nb = (N + 255) / 256;
  node_init_kernel<<<nb, 256, 0, stream>>>(x, token, Wc, xw, deg, N);
  edge_mlp_kernel<<<1024, 256, 0, stream>>>(x, token, eag, eav, gum, eidx,
      W0, b0, W1, b1, W2, b2, W3, b3, Wl, bl, ew, deg, E);
  node_norm_kernel<<<nb, 256, 0, stream>>>(deg, xw, agg, N);
  edge_agg_kernel<<<(E + 255) / 256, 256, 0, stream>>>(eidx, ew, deg, xw, agg, E);
  node_out_kernel<<<nb, 256, 0, stream>>>(agg, bc, Wo, bo, out, N);
}
